// CASSViMBlock_35622458753229
// MI455X (gfx1250) — compile-verified, matrix operand order checked by reading, not run on hardware
//
#include <hip/hip_runtime.h>
#include <hip/hip_bf16.h>

// ---------------------------------------------------------------------------
// CASS-ViM block for MI455X (gfx1250, wave32).
//   GEMMs: v_wmma_f32_16x16x32_f16, A staged by global_load_async_to_lds_b128
//   (ASYNCcnt), B staged by the Tensor Data Mover (TENSORcnt).
// ---------------------------------------------------------------------------

#define DIMC   384
#define DINNER 768
#define NB     8
#define HH     32
#define WW     32
#define LSEQ   (HH*WW)          // 1024
#define MROWS  (NB*LSEQ)        // 8192
#define DSTATE 16

typedef _Float16 v16h __attribute__((ext_vector_type(16)));
typedef float    v8f  __attribute__((ext_vector_type(8)));
typedef unsigned tdm_v4u __attribute__((ext_vector_type(4)));
typedef int      tdm_v4i __attribute__((ext_vector_type(4)));
typedef int      tdm_v8i __attribute__((ext_vector_type(8)));

union FragU { v16h h; unsigned u[8]; };

__device__ __forceinline__ unsigned pk2h(float a, float b) {
    union { _Float16 h[2]; unsigned u; } t;
    t.h[0] = (_Float16)a;
    t.h[1] = (_Float16)b;
    return t.u;
}

// ---------------------------------------------------------------------------
// Tensor Data Mover: 2-D f16 tile load (global -> LDS), built per D# spec.
//   extents are "remaining from tile start" so hardware zero-fills OOB rows.
// ---------------------------------------------------------------------------
__device__ __forceinline__ void tdm_load_2d_f16(unsigned ldsOff, const void* gptr,
                                                unsigned extent0, unsigned extent1,
                                                unsigned tile0, unsigned tile1,
                                                unsigned stride0)
{
    unsigned long long ga = (unsigned long long)(uintptr_t)gptr;
    tdm_v4u g0;
    g0[0] = 1u;                                    // count=1, user D#
    g0[1] = ldsOff;                                // lds_addr (bytes)
    g0[2] = (unsigned)(ga & 0xffffffffu);          // global_addr[31:0]
    g0[3] = (unsigned)((ga >> 32) & 0x01ffffffu)   // global_addr[56:32]
          | (2u << 30);                            // type = 2 ("image")
    tdm_v8i g1;
    g1[0] = (int)(1u << 16);                       // wg_mask=0, data_size=1 (2B)
    g1[1] = (int)((extent0 & 0xffffu) << 16);      // abar=0 | tensor_dim0 lo16
    g1[2] = (int)((extent0 >> 16) | ((extent1 & 0xffffu) << 16));
    g1[3] = (int)((extent1 >> 16) | (tile0 << 16));// tensor_dim1 hi | tile_dim0
    g1[4] = (int)(tile1 & 0xffffu);                // tile_dim1 | tile_dim2=0
    g1[5] = (int)stride0;                          // tensor_dim0_stride lo32
    g1[6] = 0;                                     // stride0 hi16 | stride1 lo16
    g1[7] = 0;
    tdm_v4i gz = {0, 0, 0, 0};
#if __clang_major__ >= 23
    tdm_v8i gz8 = {0, 0, 0, 0, 0, 0, 0, 0};
    __builtin_amdgcn_tensor_load_to_lds(g0, g1, gz, gz, gz8, 0);
#else
    __builtin_amdgcn_tensor_load_to_lds(g0, g1, gz, gz, 0);
#endif
}

// ---------------------------------------------------------------------------
// 1) LayerNorm over channels; also emits per-pixel channel-mean xg.
// ---------------------------------------------------------------------------
__global__ __launch_bounds__(128)
void ln_kernel(const float* __restrict__ x, const float* __restrict__ g,
               const float* __restrict__ be, float* __restrict__ xn,
               float* __restrict__ xg)
{
    __shared__ float r1[128], r2[128];
    const int row = blockIdx.x;
    const int tid = threadIdx.x;
    const float* xr = x + (size_t)row * DIMC;

    float s = 0.f, s2 = 0.f;
    for (int c = tid; c < DIMC; c += 128) { float v = xr[c]; s += v; s2 += v * v; }
    r1[tid] = s; r2[tid] = s2; __syncthreads();
    for (int st = 64; st > 0; st >>= 1) {
        if (tid < st) { r1[tid] += r1[tid + st]; r2[tid] += r2[tid + st]; }
        __syncthreads();
    }
    const float mu  = r1[0] * (1.f / DIMC);
    const float var = r2[0] * (1.f / DIMC) - mu * mu;
    const float inv = rsqrtf(var + 1e-5f);
    __syncthreads();

    float sg = 0.f;
    for (int c = tid; c < DIMC; c += 128) {
        float v = (xr[c] - mu) * inv * g[c] + be[c];
        xn[(size_t)row * DIMC + c] = v;
        sg += v;
    }
    r1[tid] = sg; __syncthreads();
    for (int st = 64; st > 0; st >>= 1) {
        if (tid < st) r1[tid] += r1[tid + st];
        __syncthreads();
    }
    if (tid == 0) xg[row] = r1[0] * (1.f / DIMC);
}

// ---------------------------------------------------------------------------
// 2) Direction selection MLP: one block per batch.
// ---------------------------------------------------------------------------
__device__ __forceinline__ int refl(int i) {
    return i < 0 ? -i : (i > 31 ? 62 - i : i);
}

__global__ __launch_bounds__(256)
void dir_kernel(const float* __restrict__ xg,
                const float* __restrict__ w1, const float* __restrict__ b1,
                const float* __restrict__ w2, const float* __restrict__ b2,
                int* __restrict__ idx)
{
    const int b = blockIdx.x;
    const int tid = threadIdx.x;
    __shared__ float sg[32][32];
    __shared__ float gh[34][32];
    __shared__ float gv[32][34];
    __shared__ float ghr[32][32];
    __shared__ float gvr[32][32];
    __shared__ float red[256];
    __shared__ float sres[4];

    for (int p = tid; p < 1024; p += 256)
        sg[p >> 5][p & 31] = xg[b * 1024 + p];
    __syncthreads();

    for (int p = tid; p < 34 * 32; p += 256) {
        int i = p >> 5, j = p & 31;
        int r = refl(i - 1);
        gh[i][j] = fabsf(sg[r][refl(j + 1)] - sg[r][refl(j - 1)]);
    }
    for (int p = tid; p < 32 * 34; p += 256) {
        int i = p / 34, j = p % 34;
        int cc = refl(j - 1);
        gv[i][j] = fabsf(sg[refl(i + 1)][cc] - sg[refl(i - 1)][cc]);
    }
    __syncthreads();

    for (int p = tid; p < 1024; p += 256) {
        int i = p >> 5, j = p & 31;
        float s  = (i + 0.5f) * (34.f / 32.f) - 0.5f;
        int i0 = (int)floorf(s); float fi = s - (float)i0;
        int i1 = min(i0 + 1, 33); i0 = max(i0, 0);
        ghr[i][j] = (1.f - fi) * gh[i0][j] + fi * gh[i1][j];

        float t  = (j + 0.5f) * (34.f / 32.f) - 0.5f;
        int j0 = (int)floorf(t); float fj = t - (float)j0;
        int j1 = min(j0 + 1, 33); j0 = max(j0, 0);
        gvr[i][j] = (1.f - fj) * gv[i][j0] + fj * gv[i][j1];
    }
    __syncthreads();

    float ph = 0.f, pv = 0.f, pd = 0.f, pa = 0.f;
    for (int p = tid; p < 1024; p += 256) {
        int i = p >> 5, j = p & 31;
        float sh = 0.f, sv = 0.f, sd = 0.f, sa = 0.f;
        for (int di = -1; di <= 1; ++di)
            for (int dj = -1; dj <= 1; ++dj) {
                int ii = i + di, jj = j + dj;
                if (ii >= 0 && ii < 32 && jj >= 0 && jj < 32) {
                    float hh = ghr[ii][jj], vv = gvr[ii][jj];
                    sh += hh; sv += vv;
                    sd += 0.5f * (hh + vv);
                    sa += fabsf(hh - vv);
                }
            }
        ph += sh; pv += sv; pd += sd; pa += sa;
    }
    float parts[4] = { ph, pv, pd, pa };
    for (int k = 0; k < 4; ++k) {
        __syncthreads();
        red[tid] = parts[k];
        __syncthreads();
        for (int st = 128; st > 0; st >>= 1) {
            if (tid < st) red[tid] += red[tid + st];
            __syncthreads();
        }
        if (tid == 0) sres[k] = red[0] * (1.f / (9.f * 1024.f));
    }
    __syncthreads();

    if (tid == 0) {
        float hd[32];
        for (int j = 0; j < 32; ++j) {
            float a = b1[j];
            for (int i = 0; i < 4; ++i) a += sres[i] * w1[i * 32 + j];
            hd[j] = a > 0.f ? a : 0.f;
        }
        float best = -1e30f; int bi = 0;
        for (int k = 0; k < 4; ++k) {
            float o = b2[k];
            for (int j = 0; j < 32; ++j) o += hd[j] * w2[j * 4 + k];
            if (o > best) { best = o; bi = k; }
        }
        idx[b] = bi;
    }
}

// ---------------------------------------------------------------------------
// 3) Gather sequence (optional per-batch transpose), output f16.
// ---------------------------------------------------------------------------
__global__ __launch_bounds__(256)
void gather_kernel(const float* __restrict__ xn, const int* __restrict__ idx,
                   _Float16* __restrict__ seqh)
{
    const int t = blockIdx.x * 256 + threadIdx.x;   // one float4 each
    const int total = NB * LSEQ * (DIMC / 4);
    if (t >= total) return;
    int c4 = t % (DIMC / 4);
    int l  = (t / (DIMC / 4)) & (LSEQ - 1);
    int b  = t / ((DIMC / 4) * LSEQ);
    int vert = (idx[b] == 1);
    int h = l >> 5, w = l & 31;
    int src = vert ? (w * 32 + h) : l;
    const float4 v = *(const float4*)(xn + (size_t)(b * LSEQ + src) * DIMC + c4 * 4);
    uint2 p; p.x = pk2h(v.x, v.y); p.y = pk2h(v.z, v.w);
    *(uint2*)(seqh + (size_t)t * 4) = p;
}

// ---------------------------------------------------------------------------
// 3b) Weight convert + transpose: W[K][N] f32 -> Wt[N][K] f16.
// ---------------------------------------------------------------------------
__global__ __launch_bounds__(256)
void wtrans_kernel(const float* __restrict__ W, _Float16* __restrict__ Wt,
                   int K, int N)
{
    const int t = blockIdx.x * 256 + threadIdx.x;
    if (t >= K * N) return;
    int n = t % N, k = t / N;
    Wt[(size_t)n * K + k] = (_Float16)W[t];
}

// ---------------------------------------------------------------------------
// 4) WMMA GEMM: C[M,N] = A[M,K] @ W[K,N]; A f16 row-major, Wt f16 [N][K].
//    Block tile 128x64, 8 waves (4x2), each wave 32x32 via 4x wmma 16x16x32.
//    A tile: async global->LDS copies.   B tile: TDM (wave 0).
//    MODE 0: plain   1: +bias then softplus   2: residual add (aux = x)
// ---------------------------------------------------------------------------
#define BM 128
#define BN 64
#define KC 32

template <int MODE>
__global__ __launch_bounds__(256)
void gemm_wmma(const _Float16* __restrict__ Ah, const _Float16* __restrict__ Wt,
               float* __restrict__ C, int M, int N, int K,
               const float* __restrict__ aux)
{
    __shared__ _Float16 As[BM][KC];   // 8 KB, rows K-contiguous
    __shared__ _Float16 Bs[BN][KC];   // 4 KB, rows K-contiguous

    const int tid  = threadIdx.x;
    const int lane = tid & 31;
    const int wv   = tid >> 5;
    const int wm   = wv & 3;
    const int wn   = wv >> 2;
    const int blockM = blockIdx.y * BM;
    const int blockN = blockIdx.x * BN;
    const int laneRow = lane & 15;
    const int hiq     = (lane >> 4) << 2;

    const unsigned ldsA = (unsigned)(uintptr_t)(&As[0][0]);
    const unsigned ldsB = (unsigned)(uintptr_t)(&Bs[0][0]);
    const unsigned long long abase = (unsigned long long)(uintptr_t)Ah;

    v8f acc[2][2];
    for (int mi = 0; mi < 2; ++mi)
        for (int ni = 0; ni < 2; ++ni)
            acc[mi][ni] = (v8f){0.f,0.f,0.f,0.f,0.f,0.f,0.f,0.f};

    for (int k0 = 0; k0 < K; k0 += KC) {
        // ---- A tile: 128x32 halves = 512 x 16B chunks, 2 per thread (async)
        #pragma unroll
        for (int i = 0; i < 2; ++i) {
            int c = tid + i * 256;
            int r = c >> 2;               // row 0..127
            int q = c & 3;                // 16B chunk in row
            unsigned goff = (unsigned)((blockM + r) * K + k0 + q * 8) * 2u;
            unsigned loff = ldsA + (unsigned)c * 16u;
            asm volatile("global_load_async_to_lds_b128 %0, %1, %2"
                         :: "v"(loff), "v"(goff), "s"(abase) : "memory");
        }
        // ---- B tile: TDM 2-D descriptor, one issue per block (wave 0)
        if (wv == 0) {
            tdm_load_2d_f16(ldsB, Wt + (size_t)blockN * K + k0,
                            (unsigned)(K - k0), (unsigned)(N - blockN),
                            (unsigned)KC, (unsigned)BN, (unsigned)K);
            __builtin_amdgcn_s_wait_tensorcnt(0);
        }
        asm volatile("s_wait_asynccnt 0x0" ::: "memory");
        __syncthreads();

        // ---- fragments per documented 16-bit layout (K-pairs per VGPR)
        FragU af[2], bf[2];
        #pragma unroll
        for (int s = 0; s < 2; ++s) {
            const unsigned* ap = (const unsigned*)(&As[wm * 32 + s * 16 + laneRow][0]);
            const unsigned* bp = (const unsigned*)(&Bs[wn * 32 + s * 16 + laneRow][0]);
            #pragma unroll
            for (int v = 0; v < 8; ++v) {
                int ui = ((v >> 2) << 3) | hiq | (v & 3);
                af[s].u[v] = ap[ui];
                bf[s].u[v] = bp[ui];
            }
        }
        #pragma unroll
        for (int mi = 0; mi < 2; ++mi)
            #pragma unroll
            for (int ni = 0; ni < 2; ++ni)
                acc[mi][ni] = __builtin_amdgcn_wmma_f32_16x16x32_f16(
                    false, af[mi].h, false, bf[ni].h,
                    (short)0, acc[mi][ni], false, false);
        __syncthreads();
    }

    // ---- store with epilogue (C/D layout: VGPR r -> M=r, +8 for hi lanes)
    const int mAdd = (lane >> 4) << 3;
    #pragma unroll
    for (int mi = 0; mi < 2; ++mi) {
        #pragma unroll
        for (int ni = 0; ni < 2; ++ni) {
            const int col = blockN + wn * 32 + ni * 16 + laneRow;
            if (col >= N) continue;
            #pragma unroll
            for (int r = 0; r < 8; ++r) {
                const int row = blockM + wm * 32 + mi * 16 + r + mAdd;
                float v = acc[mi][ni][r];
                size_t o = (size_t)row * N + col;
                if (MODE == 1) {
                    v += aux[col];
                    v = (v > 20.f) ? v : log1pf(__expf(v));
                } else if (MODE == 2) {
                    v += aux[o];
                }
                C[o] = v;
            }
        }
    }
}

// ---------------------------------------------------------------------------
// 5) Depthwise conv (k=3, zero pad) along L + SiLU; f32 + f16 outputs.
// ---------------------------------------------------------------------------
__global__ __launch_bounds__(256)
void conv_silu_kernel(const float* __restrict__ xz, const float* __restrict__ cw,
                      const float* __restrict__ cb, float* __restrict__ xca,
                      _Float16* __restrict__ xcah)
{
    const int t = blockIdx.x * 256 + threadIdx.x;
    const int total = NB * LSEQ * DINNER;
    if (t >= total) return;
    int d = t % DINNER;
    int l = (t / DINNER) & (LSEQ - 1);
    int b = t / (DINNER * LSEQ);
    const float* base = xz + (size_t)(b * LSEQ + l) * (2 * DINNER);
    float x1 = base[d];
    float x0 = (l > 0)        ? base[d - 2 * DINNER] : 0.f;
    float x2 = (l < LSEQ - 1) ? base[d + 2 * DINNER] : 0.f;
    float v = cb[d] + cw[d * 3 + 0] * x0 + cw[d * 3 + 1] * x1 + cw[d * 3 + 2] * x2;
    float s = v / (1.f + __expf(-v));
    xca[t]  = s;
    xcah[t] = (_Float16)s;
}

// ---------------------------------------------------------------------------
// 6) Selective scan: one thread per (b,d), 16 states in registers.
//    Fused epilogue: y = (scan + D*xc) * silu(z), stored as f16 (GEMM A).
// ---------------------------------------------------------------------------
__global__ __launch_bounds__(256)
void scan_kernel(const float* __restrict__ delta, const float* __restrict__ xca,
                 const float* __restrict__ bc,    const float* __restrict__ A_log,
                 const float* __restrict__ Dv,    const float* __restrict__ xz,
                 _Float16* __restrict__ yh)
{
    __shared__ float sB[DSTATE], sC[DSTATE];
    const int b  = blockIdx.x / (DINNER / 256);
    const int dc = blockIdx.x % (DINNER / 256);
    const int d  = dc * 256 + threadIdx.x;
    const int tid = threadIdx.x;

    float Ar[DSTATE];
    #pragma unroll
    for (int n = 0; n < DSTATE; ++n) Ar[n] = -__expf(A_log[d * DSTATE + n]);
    float h[DSTATE];
    #pragma unroll
    for (int n = 0; n < DSTATE; ++n) h[n] = 0.f;
    const float Dd = Dv[d];

    const size_t baseBC = (size_t)b * LSEQ * (2 * DSTATE);
    const size_t baseD  = (size_t)b * LSEQ * DINNER + d;
    const size_t baseZ  = (size_t)b * LSEQ * (2 * DINNER) + DINNER + d;

    for (int l = 0; l < LSEQ; ++l) {
        if (tid < 2 * DSTATE) {
            float v = bc[baseBC + (size_t)l * (2 * DSTATE) + tid];
            if (tid < DSTATE) sB[tid] = v; else sC[tid - DSTATE] = v;
        }
        __syncthreads();
        const float dl = delta[baseD + (size_t)l * DINNER];
        const float xv = xca[baseD + (size_t)l * DINNER];
        const float du = dl * xv;
        float yv = 0.f;
        #pragma unroll
        for (int n = 0; n < DSTATE; ++n) {
            float da = __expf(dl * Ar[n]);
            h[n] = da * h[n] + du * sB[n];
            yv += h[n] * sC[n];
        }
        const float z = xz[baseZ + (size_t)l * (2 * DINNER)];
        yh[baseD + (size_t)l * DINNER] =
            (_Float16)((yv + Dd * xv) * (z / (1.f + __expf(-z))));
        __syncthreads();
    }
}

// ---------------------------------------------------------------------------
// Launch
// ---------------------------------------------------------------------------
extern "C" void kernel_launch(void* const* d_in, const int* in_sizes, int n_in,
                              void* d_out, int out_size, void* d_ws, size_t ws_size,
                              hipStream_t stream)
{
    const float* x       = (const float*)d_in[0];
    const float* ln_g    = (const float*)d_in[1];
    const float* ln_b    = (const float*)d_in[2];
    const float* mlp_w1  = (const float*)d_in[3];
    const float* mlp_b1  = (const float*)d_in[4];
    const float* mlp_w2  = (const float*)d_in[5];
    const float* mlp_b2  = (const float*)d_in[6];
    const float* in_w    = (const float*)d_in[7];
    const float* conv_w  = (const float*)d_in[8];
    const float* conv_b  = (const float*)d_in[9];
    const float* xproj_w = (const float*)d_in[10];
    const float* dt_w    = (const float*)d_in[11];
    const float* dt_b    = (const float*)d_in[12];
    const float* A_log   = (const float*)d_in[13];
    const float* Dv      = (const float*)d_in[14];
    const float* out_w   = (const float*)d_in[15];
    float* out = (float*)d_out;

    // --- workspace carve-up (64B aligned blocks)
    char* wp = (char*)d_ws;
    auto alloc = [&](size_t bytes) -> char* {
        char* p = wp;
        wp += (bytes + 63) & ~(size_t)63;
        return p;
    };
    float*    xn    = (float*)   alloc((size_t)MROWS * DIMC * 4);
    float*    xg    = (float*)   alloc((size_t)MROWS * 4);
    int*      idx   = (int*)     alloc(64);
    float*    xz    = (float*)   alloc((size_t)MROWS * 2 * DINNER * 4);
    float*    xca   = (float*)   alloc((size_t)MROWS * DINNER * 4);
    float*    delta = (float*)   alloc((size_t)MROWS * DINNER * 4);
    float*    bc    = (float*)   alloc((size_t)MROWS * 2 * DSTATE * 4);
    _Float16* seqh  = (_Float16*)alloc((size_t)MROWS * DIMC * 2);
    _Float16* xcah  = (_Float16*)alloc((size_t)MROWS * DINNER * 2);
    _Float16* yh    = (_Float16*)alloc((size_t)MROWS * DINNER * 2);
    _Float16* wt_in = (_Float16*)alloc((size_t)DIMC * 2 * DINNER * 2);
    _Float16* wt_dt = (_Float16*)alloc((size_t)DINNER * DINNER * 2);
    _Float16* wt_xp = (_Float16*)alloc((size_t)DINNER * 2 * DSTATE * 2);
    _Float16* wt_out= (_Float16*)alloc((size_t)DINNER * DIMC * 2);

    // 1) layernorm + per-pixel channel mean
    ln_kernel<<<MROWS, 128, 0, stream>>>(x, ln_g, ln_b, xn, xg);

    // 2) direction-selection MLP -> idx[b]
    dir_kernel<<<NB, 256, 0, stream>>>(xg, mlp_w1, mlp_b1, mlp_w2, mlp_b2, idx);

    // 2b) weight convert + transpose to [N][K] f16
    wtrans_kernel<<<(DIMC * 2 * DINNER + 255) / 256, 256, 0, stream>>>(in_w,    wt_in,  DIMC,   2 * DINNER);
    wtrans_kernel<<<(DINNER * DINNER  + 255) / 256, 256, 0, stream>>>(dt_w,    wt_dt,  DINNER, DINNER);
    wtrans_kernel<<<(DINNER * 2 * DSTATE + 255) / 256, 256, 0, stream>>>(xproj_w, wt_xp, DINNER, 2 * DSTATE);
    wtrans_kernel<<<(DINNER * DIMC    + 255) / 256, 256, 0, stream>>>(out_w,   wt_out, DINNER, DIMC);

    // 3) gather sequence (transpose for vertical batches) -> f16
    gather_kernel<<<(NB * LSEQ * (DIMC / 4) + 255) / 256, 256, 0, stream>>>(xn, idx, seqh);

    // 4) in_proj: xz = seq @ in_w   (8192x384 @ 384x1536)
    gemm_wmma<0><<<dim3((2 * DINNER) / BN, MROWS / BM), 256, 0, stream>>>(
        seqh, wt_in, xz, MROWS, 2 * DINNER, DIMC, nullptr);

    // 5) depthwise conv + SiLU
    conv_silu_kernel<<<(NB * LSEQ * DINNER + 255) / 256, 256, 0, stream>>>(
        xz, conv_w, conv_b, xca, xcah);

    // 6) delta = softplus(xca @ dt_w + dt_b)   (8192x768 @ 768x768)
    gemm_wmma<1><<<dim3(DINNER / BN, MROWS / BM), 256, 0, stream>>>(
        xcah, wt_dt, delta, MROWS, DINNER, DINNER, dt_b);

    // 7) BC = xca @ xproj_w   (8192x768 @ 768x32, TDM zero-fills N tail)
    gemm_wmma<0><<<dim3(1, MROWS / BM), 256, 0, stream>>>(
        xcah, wt_xp, bc, MROWS, 2 * DSTATE, DINNER, nullptr);

    // 8) selective scan + gating -> f16
    scan_kernel<<<NB * (DINNER / 256), 256, 0, stream>>>(
        delta, xca, bc, A_log, Dv, xz, yh);

    // 9) out = x + yh @ out_w   (8192x768 @ 768x384, residual epilogue)
    gemm_wmma<2><<<dim3(DIMC / BN, MROWS / BM), 256, 0, stream>>>(
        yh, wt_out, out, MROWS, DIMC, DINNER, x);
}
